// SCAESuite_40896678593116
// MI455X (gfx1250) — compile-verified, matrix operand order checked by reading, not run on hardware
//
#include <hip/hip_runtime.h>

#define Bdim 2
#define Sdim 512
#define Fdim 4096
#define Ddim 768
#define Hdim 12
#define GCH  1024   // g-chunk width for the fused VW stage (16*GCH*2B = 32 KB LDS)

typedef __bf16 bf16_t;
typedef __attribute__((ext_vector_type(16))) __bf16 v16bf;
typedef __attribute__((ext_vector_type(8)))  __bf16 v8bf;
typedef __attribute__((ext_vector_type(8)))  float  v8f;

// Load a 16x32 bf16 operand tile from a row-major [rows x ld] array.
// Works for both A ([M x K]) and B stored as [N x K] row-major.
// Per ISA 16-bit A layout: lanes 0-15 hold K=0..7 & 16..23 of row (lane&15),
// lanes 16-31 hold K=8..15 & 24..31. Two 16-byte loads per lane.
__device__ __forceinline__ v16bf load_tile16x32(const bf16_t* base, int ld,
                                                int row, int kbase, int lane) {
  const bf16_t* p = base + (size_t)(row + (lane & 15)) * ld
                         + (size_t)(kbase + ((lane >> 4) << 3));
  union { v16bf v; v8bf h[2]; } u;
  u.h[0] = *(const v8bf*)(p);
  u.h[1] = *(const v8bf*)(p + 16);
  return u.v;
}

__device__ __forceinline__ v8f wmma_bf16(v16bf a, v16bf b, v8f c) {
  return __builtin_amdgcn_wmma_f32_16x16x32_bf16(false, a, false, b,
                                                 (short)0, c, false, false);
}

// ---------------- conversion kernels ----------------
__global__ void k_conv(const float* __restrict__ in, bf16_t* __restrict__ out, int n) {
  int i = blockIdx.x * blockDim.x + threadIdx.x;
  if (i < n) out[i] = (bf16_t)in[i];
}

__global__ void k_up_post(const float* __restrict__ up, const float* __restrict__ scale,
                          bf16_t* __restrict__ out, int n) {
  int i = blockIdx.x * blockDim.x + threadIdx.x;
  if (i < n) out[i] = (bf16_t)(up[i] / scale[i / Fdim]);
}

// up_decoder is [D x F]; store transposed [F x D] (N-by-K B-operand) via LDS tile
// transpose: coalesced f32 reads and coalesced bf16 writes.
__global__ __launch_bounds__(256) void k_updecT(const float* __restrict__ ud,
                                                bf16_t* __restrict__ outT) {
  __shared__ float tile[32][33];
  const int tx = threadIdx.x & 31;          // fast dim
  const int ty = threadIdx.x >> 5;          // 0..7
  const int g0 = blockIdx.x * 32;           // F / 32
  const int d0 = blockIdx.y * 32;           // D / 32
#pragma unroll
  for (int r = 0; r < 4; ++r) {             // read ud[d0+row][g0+tx]
    int row = ty + r * 8;
    tile[row][tx] = ud[(size_t)(d0 + row) * Fdim + (g0 + tx)];
  }
  __syncthreads();
#pragma unroll
  for (int r = 0; r < 4; ++r) {             // write outT[g0+row][d0+tx]
    int row = ty + r * 8;
    outT[(size_t)(g0 + row) * Ddim + (d0 + tx)] = (bf16_t)tile[tx][row];
  }
}

// ---------------- GEMM 1: tmp[h,f,i] = sum_o denc[f,o] * W_OV[h,i,o] ----------------
// A = denc [F x D] row-major (M=f,K=o); B = W_OV[h] [D x D] row-major = [N=i x K=o].
__global__ __launch_bounds__(32) void k_tmp(const bf16_t* __restrict__ dencB,
                                            const bf16_t* __restrict__ wovB,
                                            bf16_t* __restrict__ tmp) {
  const int lane = threadIdx.x & 31;
  const int f0 = blockIdx.x * 16;
  const int i0 = blockIdx.y * 16;
  const int h  = blockIdx.z;
  const bf16_t* Bm = wovB + ((size_t)h * Ddim + i0) * Ddim;
  v8f c = {};
  for (int k = 0; k < Ddim; k += 32) {
    v16bf a = load_tile16x32(dencB, Ddim, f0, k, lane);
    v16bf b = load_tile16x32(Bm, Ddim, 0, k, lane);
    c = wmma_bf16(a, b, c);
  }
#pragma unroll
  for (int r = 0; r < 8; ++r) {
    int f = f0 + r + ((lane >> 4) << 3);
    int i = i0 + (lane & 15);
    tmp[((size_t)h * Fdim + f) * Ddim + i] = (bf16_t)c[r];
  }
}

// ---------------- Fused kernel: masked virtual weights + OV application ----------------
// Per WG (h, f0..f0+15):
//   preload: tmp[h,f_tile,:]  [16 x 768] bf16 -> LDS (24 KB), 32x reuse from DS
//   stage A: VWm[16 x GCH] = (tmpA @ updecT^T) * mask  -> LDS (32 KB, bf16)
//   stage B: post_ovT accum: [1024 x 16] += up_post[1024 x g] @ VWm^T
__global__ __launch_bounds__(256) void k_vw_apply(const bf16_t* __restrict__ tmp,     // [H][F][D]
                                                  const bf16_t* __restrict__ updecT,  // [F][D]
                                                  const int*    __restrict__ mask,    // [F][F]
                                                  const bf16_t* __restrict__ up_post, // [B*S][F]
                                                  bf16_t* __restrict__ post_ovT) {    // [B][H][F][S]
  __shared__ alignas(16) bf16_t lds_vw[16 * GCH];   // 32 KB
  __shared__ alignas(16) bf16_t lds_a[16 * Ddim];   // 24 KB
  const int wg   = blockIdx.x;              // H * F/16
  const int h    = wg / (Fdim / 16);
  const int f0   = (wg % (Fdim / 16)) * 16;
  const int wave = threadIdx.x >> 5;        // 8 waves
  const int lane = threadIdx.x & 31;
  const int mbase = wave * 128;             // this wave's 128-row slab of B*S=1024

  // ---- preload A strip (contiguous [16 x 768] row-major) into LDS ----
  {
    const bf16_t* tmpA = tmp + ((size_t)h * Fdim + f0) * Ddim;
    for (int i = threadIdx.x * 8; i < 16 * Ddim; i += 256 * 8)
      *(v8bf*)(lds_a + i) = *(const v8bf*)(tmpA + i);
  }
  __syncthreads();

  v8f acc[8] = {};

  for (int gc = 0; gc < Fdim; gc += GCH) {
    // ---- stage A: build masked VW chunk in LDS ----
#pragma unroll 1
    for (int tt = 0; tt < (GCH / 16) / 8; ++tt) {      // 8 g-tiles per wave
      const int gt = (wave * 8 + tt) * 16;             // local g base
      v8f c = {};
      for (int k = 0; k < Ddim; k += 32) {
        v16bf a = load_tile16x32(lds_a, Ddim, 0, k, lane);
        v16bf b = load_tile16x32(updecT + (size_t)(gc + gt) * Ddim, Ddim, 0, k, lane);
        c = wmma_bf16(a, b, c);
      }
      const int gl = gt + (lane & 15);                 // local g column
      const int fb = (lane >> 4) << 3;
#pragma unroll
      for (int r = 0; r < 8; ++r) {
        int fl = fb + r;                               // local f row 0..15
        float v = mask[(size_t)(f0 + fl) * Fdim + (gc + gl)] ? c[r] : 0.0f;
        lds_vw[fl * GCH + gl] = (bf16_t)v;
      }
    }
    __syncthreads();

    // ---- stage B: accumulate up_post @ VWm^T ----
    for (int k = 0; k < GCH; k += 32) {
      v16bf b = load_tile16x32(lds_vw, GCH, 0, k, lane);   // shared B-tile
#pragma unroll
      for (int t = 0; t < 8; ++t) {
        v16bf a = load_tile16x32(up_post, Fdim, mbase + t * 16, gc + k, lane);
        acc[t] = wmma_bf16(a, b, acc[t]);
      }
    }
    __syncthreads();
  }

  // ---- write post_ovT[b][h][f][s] ----
  // For a fixed lane, acc[t][0..7] are 8 consecutive s at fixed f: one b128 store.
  {
    const int f = f0 + (lane & 15);
    const int khalf8 = (lane >> 4) << 3;
#pragma unroll
    for (int t = 0; t < 8; ++t) {
      int n = mbase + t * 16 + khalf8;      // first row of the 8-run, in [0,1024)
      int b = n >> 9;
      int s = n & (Sdim - 1);
      v8bf pk;
#pragma unroll
      for (int r = 0; r < 8; ++r) pk[r] = (bf16_t)acc[t][r];
      *(v8bf*)(post_ovT + ((((size_t)b * Hdim + h) * Fdim + f) * Sdim + s)) = pk;
    }
  }
}

// ---------------- GEMM 3: out[b,q,f] = sum_h sum_k probs[b,h,q,k] * post_ov[b,h,k,f] ----
// A = probs[b,h] [S x S] row-major (M=q,K=k); B = post_ovT[b,h] [F x S] = [N=f x K=k].
__global__ __launch_bounds__(32) void k_out(const bf16_t* __restrict__ probsB,   // [B][H][S][S]
                                            const bf16_t* __restrict__ post_ovT, // [B][H][F][S]
                                            float* __restrict__ out) {           // [B][S][F]
  const int lane = threadIdx.x & 31;
  const int q0 = blockIdx.x * 16;
  const int f0 = blockIdx.y * 16;
  const int b  = blockIdx.z;
  v8f c = {};
  for (int h = 0; h < Hdim; ++h) {
    const bf16_t* A  = probsB   + (((size_t)b * Hdim + h) * Sdim + q0) * Sdim;
    const bf16_t* Bm = post_ovT + (((size_t)b * Hdim + h) * Fdim + f0) * Sdim;
    for (int k = 0; k < Sdim; k += 32) {
      v16bf a = load_tile16x32(A, Sdim, 0, k, lane);
      v16bf bb = load_tile16x32(Bm, Sdim, 0, k, lane);
      c = wmma_bf16(a, bb, c);
    }
  }
#pragma unroll
  for (int r = 0; r < 8; ++r) {
    int q = q0 + r + ((lane >> 4) << 3);
    int f = f0 + (lane & 15);
    out[((size_t)b * Sdim + q) * Fdim + f] = c[r];
  }
}

// ---------------- launch ----------------
extern "C" void kernel_launch(void* const* d_in, const int* in_sizes, int n_in,
                              void* d_out, int out_size, void* d_ws, size_t ws_size,
                              hipStream_t stream) {
  (void)in_sizes; (void)n_in; (void)out_size; (void)ws_size;
  const float* up_facts   = (const float*)d_in[0];
  const float* hook_scale = (const float*)d_in[1];
  const float* probs      = (const float*)d_in[2];
  const float* down_enc   = (const float*)d_in[3];
  const float* up_dec     = (const float*)d_in[4];
  const float* w_ov       = (const float*)d_in[5];
  const int*   mask       = (const int*)d_in[6];
  float* out = (float*)d_out;

  char* ws = (char*)d_ws;
  size_t off = 0;
  auto carve = [&](size_t elems) -> bf16_t* {
    bf16_t* p = (bf16_t*)(ws + off);
    off += (elems * sizeof(bf16_t) + 255) & ~(size_t)255;
    return p;
  };
  bf16_t* up_postB = carve((size_t)Bdim * Sdim * Fdim);
  bf16_t* probsB   = carve((size_t)Bdim * Hdim * Sdim * Sdim);
  bf16_t* dencB    = carve((size_t)Fdim * Ddim);
  bf16_t* wovB     = carve((size_t)Hdim * Ddim * Ddim);
  bf16_t* updecT   = carve((size_t)Fdim * Ddim);
  bf16_t* tmpB     = carve((size_t)Hdim * Fdim * Ddim);
  bf16_t* postovT  = carve((size_t)Bdim * Hdim * Fdim * Sdim);

  int n;
  n = Bdim * Sdim * Fdim;
  k_up_post<<<(n + 255) / 256, 256, 0, stream>>>(up_facts, hook_scale, up_postB, n);
  n = Bdim * Hdim * Sdim * Sdim;
  k_conv<<<(n + 255) / 256, 256, 0, stream>>>(probs, probsB, n);
  n = Fdim * Ddim;
  k_conv<<<(n + 255) / 256, 256, 0, stream>>>(down_enc, dencB, n);
  n = Hdim * Ddim * Ddim;
  k_conv<<<(n + 255) / 256, 256, 0, stream>>>(w_ov, wovB, n);
  k_updecT<<<dim3(Fdim / 32, Ddim / 32), 256, 0, stream>>>(up_dec, updecT);

  k_tmp<<<dim3(Fdim / 16, Ddim / 16, Hdim), 32, 0, stream>>>(dencB, wovB, tmpB);
  k_vw_apply<<<Hdim * (Fdim / 16), 256, 0, stream>>>(tmpB, updecT, mask, up_postB, postovT);
  k_out<<<dim3(Sdim / 16, Fdim / 16, Bdim), 32, 0, stream>>>(probsB, postovT, out);
}
